// CausalSelfAttention_77876347011492
// MI455X (gfx1250) — compile-verified, matrix operand order checked by reading, not run on hardware
//
#include <hip/hip_runtime.h>

// Causal self-attention, T=8192, C=1024, single head, fp32 in/out.
// f16 WMMA (fp32 accumulate); flash-attention streaming; V stored transposed.
// CDNA5 paths: v_wmma_f32_16x16x32_f16 + GLOBAL_LOAD_ASYNC_TO_LDS_B128
// double-buffered software pipeline (ASYNCcnt), LDS-staged fragments.
// Workspace (88MB): xh qh kh vt yh (16M each) + Wqh Wkh Wvh Wph (2M each)

#define T_SEQ 8192
#define C_DIM 1024

typedef __attribute__((ext_vector_type(16))) _Float16 v16h;
typedef __attribute__((ext_vector_type(8)))  _Float16 v8h;
typedef __attribute__((ext_vector_type(4)))  _Float16 v4h;
typedef __attribute__((ext_vector_type(8)))  float    v8f;
typedef __attribute__((ext_vector_type(4)))  float    v4f;

union H16 { v16h v16; v8h v8[2]; };

__device__ __forceinline__ v16h make_frag(v8h lo, v8h hi) {
  H16 u; u.v8[0] = lo; u.v8[1] = hi; return u.v16;
}

// Row-major f16 fragment (A-layout; identical pattern serves NT B-operands).
// lane L -> row (L&15), K-halves (L>>4)*8 + [0,8) and +16.
__device__ __forceinline__ v16h load_frag(const _Float16* base, int ld) {
  int lane = threadIdx.x & 31;
  const _Float16* p = base + (size_t)(lane & 15) * ld + ((lane >> 4) << 3);
  v8h lo = *(const v8h*)p;
  v8h hi = *(const v8h*)(p + 16);
  return make_frag(lo, hi);
}

__device__ __forceinline__ v8f wmma16(v16h a, v16h b, v8f c) {
  return __builtin_amdgcn_wmma_f32_16x16x32_f16(false, a, false, b,
                                                (short)0, c, false, false);
}

// ---- CDNA5 async global->LDS DMA (ASYNCcnt tracked). INST_OFFSET applies to
// both LDS and global addresses, so one base covers consecutive b128 beats.
__device__ __forceinline__ unsigned lds_a(const void* p) {
  return (unsigned)(uintptr_t)p;   // generic LDS addr: low 32 bits = LDS offset
}
__device__ __forceinline__ void async_copy64B(unsigned ldst, const void* g) {
  asm volatile(
      "global_load_async_to_lds_b128 %0, %1, off\n\t"
      "global_load_async_to_lds_b128 %0, %1, off offset:16\n\t"
      "global_load_async_to_lds_b128 %0, %1, off offset:32\n\t"
      "global_load_async_to_lds_b128 %0, %1, off offset:48"
      :: "v"(ldst), "v"((unsigned long long)(uintptr_t)g) : "memory");
}
__device__ __forceinline__ void async_copy32B(unsigned ldst, const void* g) {
  asm volatile(
      "global_load_async_to_lds_b128 %0, %1, off\n\t"
      "global_load_async_to_lds_b128 %0, %1, off offset:16"
      :: "v"(ldst), "v"((unsigned long long)(uintptr_t)g) : "memory");
}
__device__ __forceinline__ void wait_async() {
  asm volatile("s_wait_asynccnt 0x0" ::: "memory");
}

// ---------------------------------------------------------------- convert
__global__ __launch_bounds__(256)
void cvt_f32_f16(const float4* __restrict__ in, v4h* __restrict__ out, int n4) {
  int i = blockIdx.x * blockDim.x + threadIdx.x;
  if (i < n4) {
    float4 f = in[i];
    v4h h; h[0] = (_Float16)f.x; h[1] = (_Float16)f.y;
    h[2] = (_Float16)f.z; h[3] = (_Float16)f.w;
    out[i] = h;
  }
}

// ---------------------------------------------------------------- GEMM NT
// out = (A(MxK) @ W(NxK)^T + bias) * scale
// OUTMODE 0: f16 row-major[ldo]  1: f16 transposed (out[col*ldo+row])  2: f32
template<int OUTMODE>
__global__ __launch_bounds__(256)
void gemm_nt(const _Float16* __restrict__ A, const _Float16* __restrict__ W,
             const float* __restrict__ bias, void* __restrict__ outp,
             int K, int ldo, float scale) {
  __shared__ _Float16 Ab[2][128][40];   // double-buffered, stride 40 halves
  __shared__ _Float16 Bb[2][128][40];
  const int tid = threadIdx.x, lane = tid & 31, w = tid >> 5;
  const int wm = w >> 2, wn = w & 3;            // 2 x 4 wave grid
  const int row0 = blockIdx.x * 128, col0 = blockIdx.y * 128;

  v8f acc[4][2] = {};

  const int lrow = tid >> 1, lpart = (tid & 1) << 4;  // 16 halves each
  const _Float16* Aptr = A + (size_t)(row0 + lrow) * K + lpart;
  const _Float16* Wptr = W + (size_t)(col0 + lrow) * K + lpart;
  const unsigned ldsA[2] = {lds_a(&Ab[0][lrow][lpart]), lds_a(&Ab[1][lrow][lpart])};
  const unsigned ldsB[2] = {lds_a(&Bb[0][lrow][lpart]), lds_a(&Bb[1][lrow][lpart])};

  // prologue: stage tile 0 into buffer 0
  async_copy32B(ldsA[0], Aptr);
  async_copy32B(ldsB[0], Wptr);

  for (int kt = 0; kt < K; kt += 32) {
    const int buf = (kt >> 5) & 1;
    wait_async();
    __syncthreads();
    if (kt + 32 < K) {                       // stage next tile into other buf
      async_copy32B(ldsA[buf ^ 1], Aptr + kt + 32);
      async_copy32B(ldsB[buf ^ 1], Wptr + kt + 32);
    }
    v16h bf0 = load_frag(&Bb[buf][wn * 32 +  0][0], 40);
    v16h bf1 = load_frag(&Bb[buf][wn * 32 + 16][0], 40);
    v16h af0 = load_frag(&Ab[buf][wm * 64 +  0][0], 40);
    v16h af1 = load_frag(&Ab[buf][wm * 64 + 16][0], 40);
    v16h af2 = load_frag(&Ab[buf][wm * 64 + 32][0], 40);
    v16h af3 = load_frag(&Ab[buf][wm * 64 + 48][0], 40);
    acc[0][0] = wmma16(af0, bf0, acc[0][0]);
    acc[0][1] = wmma16(af0, bf1, acc[0][1]);
    acc[1][0] = wmma16(af1, bf0, acc[1][0]);
    acc[1][1] = wmma16(af1, bf1, acc[1][1]);
    acc[2][0] = wmma16(af2, bf0, acc[2][0]);
    acc[2][1] = wmma16(af2, bf1, acc[2][1]);
    acc[3][0] = wmma16(af3, bf0, acc[3][0]);
    acc[3][1] = wmma16(af3, bf1, acc[3][1]);
  }

#pragma unroll
  for (int nt = 0; nt < 2; ++nt) {
    int col = col0 + wn * 32 + nt * 16 + (lane & 15);
    float bv = bias[col];
#pragma unroll
    for (int mt = 0; mt < 4; ++mt) {
      int rbase = row0 + wm * 64 + mt * 16 + ((lane >> 4) << 3);
#pragma unroll
      for (int v = 0; v < 8; ++v) {
        float val = (acc[mt][nt][v] + bv) * scale;
        int row = rbase + v;
        if (OUTMODE == 0)
          ((_Float16*)outp)[(size_t)row * ldo + col] = (_Float16)val;
        else if (OUTMODE == 1)
          ((_Float16*)outp)[(size_t)col * ldo + row] = (_Float16)val;
        else
          ((float*)outp)[(size_t)row * ldo + col] = val;
      }
    }
  }
}

// ------------------------------------------------------------- attention
// Block: 64 q-rows, 512 threads = 16 waves = 4 row-groups x 4 col-groups.
// QK^T K-dim split 4-ways over col-groups, deterministic LDS reduction.
// O fp32 16x256 per wave. Q has 1/sqrt(C) pre-folded. Staging is async DMA,
// double-buffered one stage ahead (V tile prefetched under softmax, next
// block-row's K tile prefetched under PV).
__global__ __launch_bounds__(512)
void attn_kernel(const _Float16* __restrict__ Q, const _Float16* __restrict__ Kh,
                 const _Float16* __restrict__ Vt, _Float16* __restrict__ Y) {
  __shared__ float    Sred[4][4][16][68];     // [cg][rg][row][key] pad 68
  __shared__ _Float16 Pb[64][72];             // probs f16
  __shared__ _Float16 KVb[2][4][64][72];      // double-buffered K/V tiles
  __shared__ float    mrow[64], lrow[64], arow[64];

  const int tid = threadIdx.x, lane = tid & 31, w = tid >> 5;
  const int rg = w >> 2, cg = w & 3;
  const int q0 = blockIdx.x * 64;
  const int rowbase = rg * 16;

  v8f o[16] = {};

  if (tid < 64) { mrow[tid] = -1e30f; lrow[tid] = 0.f; }

  // cooperative staging: 4 slots x 64 rows x 2 halves of 32 halves (64B)
  const int clsel  = tid >> 7;
  const int clrow  = (tid & 127) >> 1;
  const int clpart = (tid & 1) << 5;
  const unsigned ldsKV[2] = {lds_a(&KVb[0][clsel][clrow][clpart]),
                             lds_a(&KVb[1][clsel][clrow][clpart])};

  // persist Q fragments: 16 rows x 256 cols of this cg -> 8 frags (64 VGPRs)
  v16h qf[8];
#pragma unroll
  for (int f = 0; f < 8; ++f)
    qf[f] = load_frag(Q + (size_t)(q0 + rowbase) * C_DIM + cg * 256 + f * 32,
                      C_DIM);

  // stage K chunk 0 of tile j=0 into buffer 0
  async_copy64B(ldsKV[0], Kh + (size_t)clrow * C_DIM + clsel * 256 + clpart);

  for (int j = 0; j <= (int)blockIdx.x; ++j) {
    const int k0 = j * 64;
    v8f sacc[4] = {};

    // ---- S = Q K^T over this cg's 256-wide K slice, C chunks of 64
#pragma unroll
    for (int cc = 0; cc < 4; ++cc) {
      const int buf = cc & 1;
      wait_async();
      __syncthreads();
      if (cc < 3) {        // next K chunk
        async_copy64B(ldsKV[buf ^ 1],
                      Kh + (size_t)(k0 + clrow) * C_DIM + clsel * 256 +
                          (cc + 1) * 64 + clpart);
      } else {             // first V chunk (overlaps softmax phase)
        async_copy64B(ldsKV[0],
                      Vt + (size_t)(clsel * 256 + clrow) * T_SEQ + k0 + clpart);
      }
#pragma unroll
      for (int ks = 0; ks < 2; ++ks) {
        v16h qa = qf[cc * 2 + ks];
#pragma unroll
        for (int np = 0; np < 2; ++np) {
          v16h kb0 = load_frag(&KVb[buf][cg][(np * 2) * 16][ks * 32], 72);
          v16h kb1 = load_frag(&KVb[buf][cg][(np * 2 + 1) * 16][ks * 32], 72);
          sacc[np * 2]     = wmma16(qa, kb0, sacc[np * 2]);
          sacc[np * 2 + 1] = wmma16(qa, kb1, sacc[np * 2 + 1]);
        }
      }
    }

    // ---- deposit partial S (fixed slot per cg -> deterministic reduce)
#pragma unroll
    for (int nt = 0; nt < 4; ++nt)
#pragma unroll
      for (int v = 0; v < 8; ++v)
        Sred[cg][rg][((lane >> 4) << 3) + v][nt * 16 + (lane & 15)] =
            sacc[nt][v];
    __syncthreads();

    // ---- online softmax: 8 threads per row, 8 keys each
    {
      const int r = tid >> 3, seg = tid & 7, kk = seg * 8;
      v4f slo = {0.f, 0.f, 0.f, 0.f}, shi = {0.f, 0.f, 0.f, 0.f};
#pragma unroll
      for (int c = 0; c < 4; ++c) {
        slo += *(const v4f*)&Sred[c][r >> 4][r & 15][kk];
        shi += *(const v4f*)&Sred[c][r >> 4][r & 15][kk + 4];
      }
      float s[8] = {slo[0], slo[1], slo[2], slo[3],
                    shi[0], shi[1], shi[2], shi[3]};
      if (j == (int)blockIdx.x) {            // diagonal tile: causal mask
#pragma unroll
        for (int i = 0; i < 8; ++i)
          if (kk + i > r) s[i] = -1e30f;
      }
      float mx = s[0];
#pragma unroll
      for (int i = 1; i < 8; ++i) mx = fmaxf(mx, s[i]);
      mx = fmaxf(mx, __shfl_xor(mx, 1));
      mx = fmaxf(mx, __shfl_xor(mx, 2));
      mx = fmaxf(mx, __shfl_xor(mx, 4));
      float mold = mrow[r];
      float mnew = fmaxf(mold, mx);
      float psum = 0.f;
      v8h pv;
#pragma unroll
      for (int i = 0; i < 8; ++i) {
        float p = __expf(s[i] - mnew);
        psum += p;
        pv[i] = (_Float16)p;
      }
      *(v8h*)&Pb[r][kk] = pv;
      psum += __shfl_xor(psum, 1);
      psum += __shfl_xor(psum, 2);
      psum += __shfl_xor(psum, 4);
      if (seg == 0) {
        float al = __expf(mold - mnew);
        arow[r] = al;
        lrow[r] = lrow[r] * al + psum;
        mrow[r] = mnew;
      }
    }
    __syncthreads();

    // ---- rescale O by alpha (per-row)
    {
      const int hb = rowbase + ((lane >> 4) << 3);
      v4f a0 = *(const v4f*)&arow[hb];
      v4f a1 = *(const v4f*)&arow[hb + 4];
      float av[8] = {a0[0], a0[1], a0[2], a0[3], a1[0], a1[1], a1[2], a1[3]};
#pragma unroll
      for (int t = 0; t < 16; ++t)
#pragma unroll
        for (int v = 0; v < 8; ++v)
          o[t][v] *= av[v];
    }

    // ---- O += P @ V  (V staged from transposed Vt: rows are V-columns)
#pragma unroll
    for (int vc = 0; vc < 4; ++vc) {
      const int buf = vc & 1;
      wait_async();
      __syncthreads();
      if (vc < 3) {        // next V chunk
        async_copy64B(ldsKV[buf ^ 1],
                      Vt + (size_t)(clsel * 256 + (vc + 1) * 64 + clrow) * T_SEQ +
                          k0 + clpart);
      } else if (j + 1 <= (int)blockIdx.x) {  // K chunk 0 of next tile row
        async_copy64B(ldsKV[0],
                      Kh + (size_t)(k0 + 64 + clrow) * C_DIM + clsel * 256 +
                          clpart);
      }
#pragma unroll
      for (int ks = 0; ks < 2; ++ks) {
        v16h pa = load_frag(&Pb[rowbase][ks * 32], 72);
#pragma unroll
        for (int np = 0; np < 2; ++np) {
          v16h vb0 = load_frag(&KVb[buf][cg][(np * 2) * 16][ks * 32], 72);
          v16h vb1 = load_frag(&KVb[buf][cg][(np * 2 + 1) * 16][ks * 32], 72);
          o[vc * 4 + np * 2]     = wmma16(pa, vb0, o[vc * 4 + np * 2]);
          o[vc * 4 + np * 2 + 1] = wmma16(pa, vb1, o[vc * 4 + np * 2 + 1]);
        }
      }
    }
  }

  // ---- finalize: O /= l, write f16
  {
    const int hb = rowbase + ((lane >> 4) << 3);
    float li[8];
#pragma unroll
    for (int v = 0; v < 8; ++v) li[v] = 1.0f / lrow[hb + v];
#pragma unroll
    for (int t = 0; t < 16; ++t) {
      int col = cg * 256 + t * 16 + (lane & 15);
#pragma unroll
      for (int v = 0; v < 8; ++v) {
        int row = q0 + hb + v;
        Y[(size_t)row * C_DIM + col] = (_Float16)(o[t][v] * li[v]);
      }
    }
  }
}

// ---------------------------------------------------------------- launch
extern "C" void kernel_launch(void* const* d_in, const int* in_sizes, int n_in,
                              void* d_out, int out_size, void* d_ws, size_t ws_size,
                              hipStream_t stream) {
  const float* x  = (const float*)d_in[0];
  const float* Wq = (const float*)d_in[1];
  const float* bq = (const float*)d_in[2];
  const float* Wk = (const float*)d_in[3];
  const float* bk = (const float*)d_in[4];
  const float* Wv = (const float*)d_in[5];
  const float* bv = (const float*)d_in[6];
  const float* Wp = (const float*)d_in[7];
  const float* bp = (const float*)d_in[8];

  char* ws = (char*)d_ws;
  const size_t TC2 = (size_t)T_SEQ * C_DIM * 2;  // 16MB
  const size_t CC2 = (size_t)C_DIM * C_DIM * 2;  //  2MB
  _Float16* xh  = (_Float16*)(ws);
  _Float16* qh  = (_Float16*)(ws + TC2);
  _Float16* kh  = (_Float16*)(ws + 2 * TC2);
  _Float16* vt  = (_Float16*)(ws + 3 * TC2);   // transposed (C, T)
  _Float16* yh  = (_Float16*)(ws + 4 * TC2);
  _Float16* wqh = (_Float16*)(ws + 5 * TC2);
  _Float16* wkh = (_Float16*)(ws + 5 * TC2 + CC2);
  _Float16* wvh = (_Float16*)(ws + 5 * TC2 + 2 * CC2);
  _Float16* wph = (_Float16*)(ws + 5 * TC2 + 3 * CC2);

  {
    int n4 = T_SEQ * C_DIM / 4;
    cvt_f32_f16<<<n4 / 256, 256, 0, stream>>>((const float4*)x, (v4h*)xh, n4);
    int m4 = C_DIM * C_DIM / 4;
    cvt_f32_f16<<<m4 / 256, 256, 0, stream>>>((const float4*)Wq, (v4h*)wqh, m4);
    cvt_f32_f16<<<m4 / 256, 256, 0, stream>>>((const float4*)Wk, (v4h*)wkh, m4);
    cvt_f32_f16<<<m4 / 256, 256, 0, stream>>>((const float4*)Wv, (v4h*)wvh, m4);
    cvt_f32_f16<<<m4 / 256, 256, 0, stream>>>((const float4*)Wp, (v4h*)wph, m4);
  }

  dim3 ggrid(T_SEQ / 128, C_DIM / 128);
  const float qscale = 0.03125f;  // 1/sqrt(1024)

  gemm_nt<0><<<ggrid, 256, 0, stream>>>(xh, wqh, bq, qh, C_DIM, C_DIM, qscale);
  gemm_nt<0><<<ggrid, 256, 0, stream>>>(xh, wkh, bk, kh, C_DIM, C_DIM, 1.0f);
  gemm_nt<1><<<ggrid, 256, 0, stream>>>(xh, wvh, bv, vt, C_DIM, T_SEQ, 1.0f);

  attn_kernel<<<T_SEQ / 64, 512, 0, stream>>>(qh, kh, vt, yh);

  gemm_nt<2><<<ggrid, 256, 0, stream>>>(yh, wph, bp, d_out, C_DIM, C_DIM, 1.0f);
}
